// GAT_17334488006783
// MI455X (gfx1250) — matrix operand order repaired and checked by reading, not compiled, not run on hardware
//
#include <hip/hip_runtime.h>
#include <hip/hip_bf16.h>

// GAT 2-layer forward for MI455X (gfx1250, wave32, WMMA).
// Dominant cost: streaming x (400MB) once -> ~17us at 23.3 TB/s. GEMM1 runs
// f16 WMMA (16x16x32, f32 accum) so compute stays under the HBM roofline.
// v2: 64 rows/block (halves W1 L2 re-read traffic) + software-pipelined
// k-chunks (global loads for chunk k+1 issued during WMMA compute of chunk k).

#define NN 10000
#define EE 320000
#define E2 (EE + NN)
#define NEG_SLOPE 0.2f

typedef _Float16 f16;
typedef __attribute__((ext_vector_type(4)))  _Float16 v4h;
typedef __attribute__((ext_vector_type(8)))  _Float16 v8h;
typedef __attribute__((ext_vector_type(16))) _Float16 v16h;
typedef __attribute__((ext_vector_type(8)))  float    v8f;

__device__ __forceinline__ v8f wmma16x16x32(v16h a, v16h b, v8f c) {
#if __has_builtin(__builtin_amdgcn_wmma_f32_16x16x32_f16)
  return __builtin_amdgcn_wmma_f32_16x16x32_f16(false, a, false, b, (short)0, c,
                                                false, false);
#else
  asm volatile("v_wmma_f32_16x16x32_f16 %0, %1, %2, %0" : "+v"(c) : "v"(a), "v"(b));
  return c;
#endif
}

__device__ __forceinline__ void atomicMaxF(float* addr, float val) {
  // sign-split float max via integer atomics (init value must be -inf)
  if (val >= 0.f) atomicMax((int*)addr, __float_as_int(val));
  else            atomicMin((unsigned int*)addr, __float_as_uint(val));
}

// ---------------------------------------------------------------------------
// GEMM1: h1[N,128] = x[N,N] @ W1[N,128]; f16 WMMA, f32 accumulate.
// Block = 256 threads (8 waves) covering 64 rows x 128 cols:
//   wave = (rowTile 0..3) x (colHalf 0..1), each wave 16 rows x 64 cols.
// LDS: sA [64 rows][64 K] f16 (stride 72), sB [128 cols][64 K] f16 (stride 72,
// transposed so B fragments are contiguous). 27.6 KB of 320 KB/WGP.
// Pipeline: regs(chunk k+1) loaded while WMMAs consume LDS(chunk k).
// ---------------------------------------------------------------------------
__global__ __launch_bounds__(256) void gat_gemm1(const float* __restrict__ x,
                                                 const float* __restrict__ W1,
                                                 float* __restrict__ h1) {
  __shared__ __align__(16) f16 sA[64 * 72];
  __shared__ __align__(16) f16 sB[128 * 72];

  const int tid  = threadIdx.x;
  const int lane = tid & 31;
  const int wave = tid >> 5;
  const int hh   = lane >> 4;   // half-wave (0/1)
  const int l16  = lane & 15;
  const int rowTile = wave >> 1;   // 0..3
  const int colHalf = wave & 1;    // 0..1
  const long long blockRow0 = (long long)blockIdx.x * 64;
  const int m0 = (int)blockRow0 + rowTile * 16;

  v8f acc[4];
  const v8f zero = {0.f, 0.f, 0.f, 0.f, 0.f, 0.f, 0.f, 0.f};
#pragma unroll
  for (int i = 0; i < 4; ++i) acc[i] = zero;

  // staging roles (256 threads)
  const int ar = tid >> 2, aq = tid & 3;  // A: row 0..63, K-quarter 0..3
  const int br = tid >> 2, bq = tid & 3;  // B: K-row 0..63, col-quarter 0..3

  const long long growA = blockRow0 + ar;
  const float*    xrow  = x + growA * (long long)NN;

  float4 rA[4];  // A stage: 4x float4 = K quarter (16 values)
  float4 rB[8];  // B stage: 8x float4 = 32 cols of one W1 row

  const float4 f4z = make_float4(0.f, 0.f, 0.f, 0.f);

  // ---- prologue: stage chunk 0 into registers ----
#pragma unroll
  for (int i = 0; i < 4; ++i) {
    int kg = aq * 16 + i * 4;
    rA[i] = f4z;
    if (growA < NN && kg < NN) rA[i] = *(const float4*)(xrow + kg);
  }
  {
    const float* wrow = W1 + (long long)br * 128;
#pragma unroll
    for (int i = 0; i < 8; ++i) rB[i] = *(const float4*)(wrow + bq * 32 + i * 4);
  }

  for (int k0 = 0; k0 < NN; k0 += 64) {
    __syncthreads();  // previous compute done reading LDS
    // ---- store staged regs -> LDS (f32 -> f16 convert) ----
#pragma unroll
    for (int i = 0; i < 4; ++i) {
      v4h hv = {(f16)rA[i].x, (f16)rA[i].y, (f16)rA[i].z, (f16)rA[i].w};
      *(v4h*)(&sA[ar * 72 + aq * 16 + i * 4]) = hv;
    }
#pragma unroll
    for (int i = 0; i < 8; ++i) {
      int c = bq * 32 + i * 4;
      sB[(c + 0) * 72 + br] = (f16)rB[i].x;
      sB[(c + 1) * 72 + br] = (f16)rB[i].y;
      sB[(c + 2) * 72 + br] = (f16)rB[i].z;
      sB[(c + 3) * 72 + br] = (f16)rB[i].w;
    }
    __syncthreads();  // LDS tile ready

    // ---- issue global loads for next chunk (overlap with WMMAs below) ----
    int kn = k0 + 64;
    if (kn < NN) {
#pragma unroll
      for (int i = 0; i < 4; ++i) {
        int kg = kn + aq * 16 + i * 4;
        rA[i] = f4z;
        if (growA < NN && kg < NN) rA[i] = *(const float4*)(xrow + kg);
      }
      int kgb = kn + br;
      const float* wrow = W1 + (long long)kgb * 128;
#pragma unroll
      for (int i = 0; i < 8; ++i) {
        rB[i] = f4z;
        if (kgb < NN) rB[i] = *(const float4*)(wrow + bq * 32 + i * 4);
      }
      if (growA < NN && (kn + 64) < NN)
        __builtin_prefetch(xrow + kn + 64 + aq * 16, 0, 0);
    }

    // ---- compute: two K=32 WMMA steps over current LDS tile ----
#pragma unroll
    for (int kk = 0; kk < 64; kk += 32) {
      const f16* arow = &sA[(rowTile * 16 + l16) * 72 + kk + 16 * hh];
      v8h alo = *(const v8h*)(arow);        // K = kk+16h .. +7   (elems 0..7)
      v8h ahi = *(const v8h*)(arow + 8);    // K = kk+16h+8 .. +7 (elems 8..15)
      v16h a = __builtin_shufflevector(alo, ahi, 0, 1, 2, 3, 4, 5, 6, 7,
                                       8, 9, 10, 11, 12, 13, 14, 15);
#pragma unroll
      for (int nt = 0; nt < 4; ++nt) {
        int col = colHalf * 64 + nt * 16 + l16;
        v16h b = *(const v16h*)(&sB[col * 72 + kk + 16 * hh]);  // K=kk+16h..+15
        acc[nt] = wmma16x16x32(a, b, acc[nt]);
      }
    }
  }

  // ---- store C/D: VGPR v -> row m0 + v + 8*hh, col = colHalf*64 + nt*16 + l16 ----
#pragma unroll
  for (int nt = 0; nt < 4; ++nt) {
#pragma unroll
    for (int v = 0; v < 8; ++v) {
      int row = m0 + v + 8 * hh;
      int col = colHalf * 64 + nt * 16 + l16;
      if (row < NN) h1[(long long)row * 128 + col] = acc[nt][v];
    }
  }
}

// ---------------------------------------------------------------------------
// GEMM2: h2[N,16] = relu(out1[N,128]) @ W2[128,16]; ReLU fused into A loads.
// One 16x16 tile per wave, K = 128 in 4 WMMA steps. W2 staged transposed.
// ---------------------------------------------------------------------------
__global__ __launch_bounds__(128) void gat_gemm2(const float* __restrict__ out1,
                                                 const float* __restrict__ W2,
                                                 float* __restrict__ h2) {
  __shared__ __align__(16) f16 sB[16 * 136];  // [col][K], stride 136
  const int tid = threadIdx.x;
  {
    const float* wrow = W2 + tid * 16;  // thread t owns K row t
#pragma unroll
    for (int c = 0; c < 16; ++c) sB[c * 136 + tid] = (f16)wrow[c];
  }
  __syncthreads();

  const int lane = tid & 31, wave = tid >> 5;
  const int hh = lane >> 4, l16 = lane & 15;
  int mt = blockIdx.x * 4 + wave;
  if (mt >= NN / 16) return;  // wave-uniform exit, no barriers after
  int m0 = mt * 16;

  v8f acc = {0.f, 0.f, 0.f, 0.f, 0.f, 0.f, 0.f, 0.f};
  const float* arow = out1 + (long long)(m0 + l16) * 128;
#pragma unroll
  for (int kk = 0; kk < 128; kk += 32) {
    int ko = kk + 16 * hh;
    float4 f0 = *(const float4*)(arow + ko);
    float4 f1 = *(const float4*)(arow + ko + 4);
    float4 g0 = *(const float4*)(arow + ko + 8);
    float4 g1 = *(const float4*)(arow + ko + 12);
    v16h a = {(f16)fmaxf(f0.x, 0.f), (f16)fmaxf(f0.y, 0.f),
              (f16)fmaxf(f0.z, 0.f), (f16)fmaxf(f0.w, 0.f),
              (f16)fmaxf(f1.x, 0.f), (f16)fmaxf(f1.y, 0.f),
              (f16)fmaxf(f1.z, 0.f), (f16)fmaxf(f1.w, 0.f),
              (f16)fmaxf(g0.x, 0.f), (f16)fmaxf(g0.y, 0.f),
              (f16)fmaxf(g0.z, 0.f), (f16)fmaxf(g0.w, 0.f),
              (f16)fmaxf(g1.x, 0.f), (f16)fmaxf(g1.y, 0.f),
              (f16)fmaxf(g1.z, 0.f), (f16)fmaxf(g1.w, 0.f)};
    v16h b = *(const v16h*)(&sB[l16 * 136 + kk + 16 * hh]);
    acc = wmma16x16x32(a, b, acc);
  }
#pragma unroll
  for (int v = 0; v < 8; ++v)
    h2[(long long)(m0 + v + 8 * hh) * 16 + l16] = acc[v];
}

// ---------------------------------------------------------------------------
// Small kernels: attention dots, edge softmax pieces, scatter aggregate.
// ---------------------------------------------------------------------------
template <int C>
__global__ void gat_att_dot(const float* __restrict__ h,
                            const float* __restrict__ atts,
                            const float* __restrict__ attd,
                            float* __restrict__ oas, float* __restrict__ oad) {
  int n = blockIdx.x * blockDim.x + threadIdx.x;
  if (n >= NN) return;
  const float* row = h + (long long)n * C;
  float s = 0.f, d = 0.f;
#pragma unroll 4
  for (int c = 0; c < C; c += 4) {
    float4 v = *(const float4*)(row + c);
    float4 a = *(const float4*)(atts + c);
    float4 b = *(const float4*)(attd + c);
    s += v.x * a.x + v.y * a.y + v.z * a.z + v.w * a.w;
    d += v.x * b.x + v.y * b.y + v.z * b.z + v.w * b.w;
  }
  oas[n] = s;
  oad[n] = d;
}

__global__ void gat_init(float* __restrict__ m, float* __restrict__ denom,
                         float* __restrict__ out, const float* __restrict__ bias,
                         int C, long long total) {
  long long t = (long long)blockIdx.x * blockDim.x + threadIdx.x;
  if (t < NN) { m[t] = -__builtin_inff(); denom[t] = 0.f; }
  if (t < total) out[t] = bias[(int)(t % C)];
}

__global__ void gat_edge_logit(const int* __restrict__ ei,
                               const float* __restrict__ as,
                               const float* __restrict__ ad,
                               float* __restrict__ ex, float* __restrict__ m) {
  int e = blockIdx.x * blockDim.x + threadIdx.x;
  if (e >= E2) return;
  int s, d;
  if (e < EE) { s = ei[e]; d = ei[EE + e]; } else { s = d = e - EE; }
  float v = as[s] + ad[d];
  v = (v > 0.f) ? v : NEG_SLOPE * v;
  ex[e] = v;
  atomicMaxF(&m[d], v);
}

__global__ void gat_edge_exp(const int* __restrict__ ei, float* __restrict__ ex,
                             const float* __restrict__ m,
                             float* __restrict__ denom) {
  int e = blockIdx.x * blockDim.x + threadIdx.x;
  if (e >= E2) return;
  int d = (e < EE) ? ei[EE + e] : (e - EE);
  float v = __expf(ex[e] - m[d]);
  ex[e] = v;
  atomicAdd(&denom[d], v);
}

__global__ void gat_edge_norm(const int* __restrict__ ei, float* __restrict__ ex,
                              const float* __restrict__ denom) {
  int e = blockIdx.x * blockDim.x + threadIdx.x;
  if (e >= E2) return;
  int d = (e < EE) ? ei[EE + e] : (e - EE);
  ex[e] = ex[e] / denom[d];
}

template <int C>
__global__ void gat_aggregate(const int* __restrict__ ei,
                              const float* __restrict__ h,
                              const float* __restrict__ alpha,
                              float* __restrict__ out) {
  constexpr int LOG = (C == 128) ? 7 : 4;
  long long t = (long long)blockIdx.x * blockDim.x + threadIdx.x;
  if (t >= (long long)E2 * C) return;
  int c = (int)(t & (C - 1));
  long long e = t >> LOG;
  int s, d;
  if (e < EE) { s = ei[e]; d = ei[EE + e]; } else { s = d = (int)(e - EE); }
  atomicAdd(&out[(long long)d * C + c], h[(long long)s * C + c] * alpha[e]);
}

// ---------------------------------------------------------------------------
extern "C" void kernel_launch(void* const* d_in, const int* in_sizes, int n_in,
                              void* d_out, int out_size, void* d_ws,
                              size_t ws_size, hipStream_t stream) {
  (void)in_sizes; (void)n_in; (void)out_size; (void)ws_size;
  const float* x   = (const float*)d_in[0];
  const int*   ei  = (const int*)d_in[1];   // edge_index [2,E] flat
  const float* W1  = (const float*)d_in[3];
  const float* as1 = (const float*)d_in[4];
  const float* ad1 = (const float*)d_in[5];
  const float* b1  = (const float*)d_in[6];
  const float* W2  = (const float*)d_in[7];
  const float* as2 = (const float*)d_in[8];
  const float* ad2 = (const float*)d_in[9];
  const float* b2  = (const float*)d_in[10];
  float* out = (float*)d_out;

  float* ws   = (float*)d_ws;
  float* h1   = ws;                           // N*128
  float* out1 = h1 + (size_t)NN * 128;        // N*128
  float* h2   = out1 + (size_t)NN * 128;      // N*16
  float* aS   = h2 + (size_t)NN * 16;         // N
  float* aD   = aS + NN;                      // N
  float* mM   = aD + NN;                      // N
  float* dnm  = mM + NN;                      // N
  float* ex   = dnm + NN;                     // E2

  const int T = 256;
  const int gE = (E2 + T - 1) / T;

  // -------- layer 1 --------
  {
    long long tot = (long long)NN * 128;
    gat_init<<<(int)((tot + T - 1) / T), T, 0, stream>>>(mM, dnm, out1, b1, 128, tot);
  }
  gat_gemm1<<<(NN + 63) / 64, 256, 0, stream>>>(x, W1, h1);
  gat_att_dot<128><<<(NN + T - 1) / T, T, 0, stream>>>(h1, as1, ad1, aS, aD);
  gat_edge_logit<<<gE, T, 0, stream>>>(ei, aS, aD, ex, mM);
  gat_edge_exp<<<gE, T, 0, stream>>>(ei, ex, mM, dnm);
  gat_edge_norm<<<gE, T, 0, stream>>>(ei, ex, dnm);
  {
    long long tot = (long long)E2 * 128;
    gat_aggregate<128><<<(int)((tot + T - 1) / T), T, 0, stream>>>(ei, h1, ex, out1);
  }

  // -------- layer 2 --------
  {
    long long tot = (long long)NN * 16;
    gat_init<<<(int)((tot + T - 1) / T), T, 0, stream>>>(mM, dnm, out, b2, 16, tot);
  }
  gat_gemm2<<<((NN / 16) + 3) / 4, 128, 0, stream>>>(out1, W2, h2);
  gat_att_dot<16><<<(NN + T - 1) / T, T, 0, stream>>>(h2, as2, ad2, aS, aD);
  gat_edge_logit<<<gE, T, 0, stream>>>(ei, aS, aD, ex, mM);
  gat_edge_exp<<<gE, T, 0, stream>>>(ei, ex, mM, dnm);
  gat_edge_norm<<<gE, T, 0, stream>>>(ei, ex, dnm);
  {
    long long tot = (long long)E2 * 16;
    gat_aggregate<16><<<(int)((tot + T - 1) / T), T, 0, stream>>>(ei, h2, ex, out);
  }
}
